// CoarseFineKGModel_14027363189378
// MI455X (gfx1250) — matrix operand order repaired and checked by reading, not compiled, not run on hardware
//
#include <hip/hip_runtime.h>
#include <hip/hip_bf16.h>

typedef __attribute__((ext_vector_type(16))) _Float16 v16h;
typedef __attribute__((ext_vector_type(8)))  float    v8f;

#define N_ENT   64368
#define N_RELS  46
#define NBASES  8
#define DIM     128
#define NEDGES  600000
#define BUSERS  64
#define LCTX    50
#define NEIDS   256
#define MAXC    3456   // max unique needed entities: 64*50 + 256

// ---------------- init / index-map construction ----------------

__global__ void k_init(int* __restrict__ idx_map, int* __restrict__ counter,
                       float* __restrict__ deg, float* __restrict__ agg) {
    int i = blockIdx.x * blockDim.x + threadIdx.x;
    if (i < N_ENT) idx_map[i] = -1;
    if (i < MAXC * N_RELS) deg[i] = 0.0f;
    if (i < MAXC * DIM) agg[i] = 0.0f;
    if (i == 0) *counter = 0;
}

__global__ void k_mark(const int* __restrict__ ctx, const int* __restrict__ eids,
                       int* __restrict__ idx_map) {
    int i = blockIdx.x * blockDim.x + threadIdx.x;
    const int n = BUSERS * LCTX + NEIDS;
    if (i >= n) return;
    int e = (i < BUSERS * LCTX) ? ctx[i] : eids[i - BUSERS * LCTX];
    atomicCAS(&idx_map[e], -1, -2);
}

__global__ void k_assign(int* __restrict__ idx_map, int* __restrict__ counter) {
    int i = blockIdx.x * blockDim.x + threadIdx.x;
    if (i < N_ENT && idx_map[i] == -2) idx_map[i] = atomicAdd(counter, 1);
}

// ---------------- RGCN: filtered degree + message scatter ----------------

__global__ void k_deg(const int* __restrict__ ei, const int* __restrict__ et,
                      const int* __restrict__ idx_map, float* __restrict__ deg) {
    int e = blockIdx.x * blockDim.x + threadIdx.x;
    if (e >= NEDGES) return;
    int dst = ei[NEDGES + e];
    int c = idx_map[dst];
    if (c >= 0) atomicAdd(&deg[c * N_RELS + et[e]], 1.0f);
}

// one wave (32 lanes) per edge; each lane owns 4 consecutive floats of D=128
__global__ void k_msg(const int* __restrict__ ei, const int* __restrict__ et,
                      const int* __restrict__ idx_map, const float* __restrict__ comp,
                      const float* __restrict__ basis, const float* __restrict__ deg,
                      float* __restrict__ agg) {
    int wave = (blockIdx.x * blockDim.x + threadIdx.x) >> 5;
    int lane = threadIdx.x & 31;
    if (wave >= NEDGES) return;
    int dst = ei[NEDGES + wave];
    int c = idx_map[dst];
    if (c < 0) return;                      // wave-uniform filter
    int src = ei[wave];
    int rel = et[wave];
    float norm = 1.0f / fmaxf(deg[c * N_RELS + rel], 1.0f);
    float cf[NBASES];
#pragma unroll
    for (int b = 0; b < NBASES; b++) cf[b] = comp[rel * NBASES + b];
    float4 acc = {0.f, 0.f, 0.f, 0.f};
#pragma unroll
    for (int b = 0; b < NBASES; b++) {
        const float4* bp = (const float4*)(basis + ((size_t)b * N_ENT + src) * DIM);
        float4 v = bp[lane];
        acc.x += cf[b] * v.x; acc.y += cf[b] * v.y;
        acc.z += cf[b] * v.z; acc.w += cf[b] * v.w;
    }
    float* ap = agg + (size_t)c * DIM + lane * 4;
    atomicAdd(ap + 0, acc.x * norm);
    atomicAdd(ap + 1, acc.y * norm);
    atomicAdd(ap + 2, acc.z * norm);
    atomicAdd(ap + 3, acc.w * norm);
}

__global__ void k_finish(const int* __restrict__ idx_map, const float* __restrict__ root,
                         const float* __restrict__ bias, float* __restrict__ agg) {
    int t = blockIdx.x * blockDim.x + threadIdx.x;
    int i = t / DIM, d = t % DIM;
    if (i >= N_ENT) return;
    int c = idx_map[i];
    if (c < 0) return;
    agg[(size_t)c * DIM + d] += root[(size_t)i * DIM + d] + bias[d];
}

// ---------------- gathers ----------------

__global__ void k_gather(const int* __restrict__ ids, const int* __restrict__ idx_map,
                         const float* __restrict__ kg, float* __restrict__ out, int nrows) {
    int t = blockIdx.x * blockDim.x + threadIdx.x;
    int r = t / DIM, d = t % DIM;
    if (r >= nrows) return;
    int c = idx_map[ids[r]];
    out[(size_t)r * DIM + d] = (c >= 0) ? kg[(size_t)c * DIM + d] : 0.0f;
}

// ---------------- WMMA GEMM: out[M,128] = act(X[M,128] @ W[128,128] + bias) ----------------
// One wave per 16x16 tile; K stepped by 32 via v_wmma_f32_16x16x32_f16 (fp32 accum).
// act: 0=none, 1=relu, 2=tanh

__global__ void k_gemm_wmma(const float* __restrict__ X, const float* __restrict__ W,
                            const float* __restrict__ bias, float* __restrict__ out,
                            int M, int act) {
    int lane = threadIdx.x & 31;
    int wave = (blockIdx.x * blockDim.x + threadIdx.x) >> 5;
    int ntiles = (M >> 4) * (DIM >> 4);
    if (wave >= ntiles) return;             // wave-uniform -> EXEC all-ones for WMMA
    int tm = wave >> 3;                     // 8 column tiles (N=128)
    int tn = wave & 7;
    int l15 = lane & 15;
    int hi  = (lane >> 4) & 1;
    int rowA = tm * 16 + l15;               // A: lane indexes M
    int colB = tn * 16 + l15;               // B: lane indexes N
    v8f c = {};
#pragma unroll
    for (int k0 = 0; k0 < DIM; k0 += 32) {
        v16h a, b;
        int kbA = k0 + hi * 8;              // A halves: K 0-7 / 8-15 (+16 for elems 8..15)
        int kbB = k0 + hi * 8;              // B halves: K 0-7 / 8-15 (+16 for elems 8..15)
#pragma unroll
        for (int e = 0; e < 16; e++) {
            int ka = kbA + e + ((e >= 8) ? 8 : 0);
            a[e] = (_Float16)X[(size_t)rowA * DIM + ka];
            b[e] = (_Float16)W[(size_t)(kbB + e + ((e >= 8) ? 8 : 0)) * DIM + colB];
        }
        c = __builtin_amdgcn_wmma_f32_16x16x32_f16(false, a, false, b,
                                                   (short)0, c, false, false);
    }
    float bv = bias ? bias[colB] : 0.0f;
#pragma unroll
    for (int r = 0; r < 8; r++) {
        int mrow = tm * 16 + r + hi * 8;
        float v = c[r] + bv;
        if (act == 1) v = fmaxf(v, 0.0f);
        else if (act == 2) v = tanhf(v);
        out[(size_t)mrow * DIM + colB] = v;
    }
}

// ---------------- attention: scores + masked softmax pool ----------------

__global__ void k_scores(const float* __restrict__ t, const float* __restrict__ attn_b,
                         float* __restrict__ scores) {
    int lane = threadIdx.x & 31;
    int row = (blockIdx.x * blockDim.x + threadIdx.x) >> 5;
    if (row >= BUSERS * LCTX) return;
    float s = 0.0f;
    for (int d = lane; d < DIM; d += 32) s += t[(size_t)row * DIM + d] * attn_b[d];
    for (int o = 16; o > 0; o >>= 1) s += __shfl_down(s, o, 32);
    if (lane == 0) scores[row] = s;
}

__global__ void k_attnpool(const float* __restrict__ scores,
                           const unsigned char* __restrict__ mask,
                           const float* __restrict__ h, float* __restrict__ urep) {
    __shared__ float s_sc[LCTX];
    __shared__ float s_keep[LCTX];
    int b = blockIdx.x, d = threadIdx.x;
    if (d < LCTX) {
        float keep = mask[b * LCTX + d] ? 1.0f : 0.0f;
        s_keep[d] = keep;
        s_sc[d] = keep ? scores[b * LCTX + d] : -1e9f;
    }
    __syncthreads();
    float mx = -1e30f;
    for (int l = 0; l < LCTX; l++) mx = fmaxf(mx, s_sc[l]);
    float sum = 0.0f;
    for (int l = 0; l < LCTX; l++) sum += __expf(s_sc[l] - mx);
    float inv = 1.0f / sum;
    float acc = 0.0f;
    for (int l = 0; l < LCTX; l++) {
        float w = __expf(s_sc[l] - mx) * inv * s_keep[l];
        acc += w * h[(size_t)(b * LCTX + l) * DIM + d];
    }
    urep[b * DIM + d] = acc;
}

// ---------------- launch ----------------

extern "C" void kernel_launch(void* const* d_in, const int* in_sizes, int n_in,
                              void* d_out, int out_size, void* d_ws, size_t ws_size,
                              hipStream_t stream) {
    const int*   edge_index = (const int*)d_in[0];
    const int*   edge_type  = (const int*)d_in[1];
    const int*   ctx        = (const int*)d_in[2];
    const unsigned char* mask = (const unsigned char*)d_in[3];
    const int*   eids       = (const int*)d_in[4];
    const float* comp   = (const float*)d_in[5];
    const float* basis  = (const float*)d_in[6];
    const float* root   = (const float*)d_in[7];
    const float* bias   = (const float*)d_in[8];
    const float* attn_W = (const float*)d_in[9];
    const float* attn_b = (const float*)d_in[10];
    const float* fc1_W  = (const float*)d_in[11];
    const float* fc1_b  = (const float*)d_in[12];
    const float* fc2_W  = (const float*)d_in[13];
    const float* fc2_b  = (const float*)d_in[14];
    const float* efc1_W = (const float*)d_in[15];
    const float* efc1_b = (const float*)d_in[16];
    const float* efc2_W = (const float*)d_in[17];
    const float* efc2_b = (const float*)d_in[18];

    char* ws = (char*)d_ws;
    size_t off = 0;
    auto alloc = [&](size_t bytes) -> void* {
        void* p = ws + off;
        off = (off + bytes + 255) & ~(size_t)255;
        return p;
    };
    int*   idx_map = (int*)  alloc((size_t)N_ENT * 4);
    int*   counter = (int*)  alloc(256);
    float* deg     = (float*)alloc((size_t)MAXC * N_RELS * 4);
    float* agg     = (float*)alloc((size_t)MAXC * DIM * 4);
    float* hbuf    = (float*)alloc((size_t)BUSERS * LCTX * DIM * 4);
    float* tbuf    = (float*)alloc((size_t)BUSERS * LCTX * DIM * 4);
    float* scores  = (float*)alloc((size_t)BUSERS * LCTX * 4);
    float* urep    = (float*)alloc((size_t)BUSERS * DIM * 4);
    float* hidden  = (float*)alloc((size_t)NEIDS * DIM * 4);
    float* ereps   = (float*)alloc((size_t)NEIDS * DIM * 4);

    float* out_user = (float*)d_out;                 // [64,128]
    float* out_ent  = out_user + BUSERS * DIM;       // [256,128]

    const int initN = MAXC * DIM; // largest of the init regions
    k_init  <<<(initN + 255) / 256, 256, 0, stream>>>(idx_map, counter, deg, agg);
    k_mark  <<<(BUSERS * LCTX + NEIDS + 255) / 256, 256, 0, stream>>>(ctx, eids, idx_map);
    k_assign<<<(N_ENT + 255) / 256, 256, 0, stream>>>(idx_map, counter);
    k_deg   <<<(NEDGES + 255) / 256, 256, 0, stream>>>(edge_index, edge_type, idx_map, deg);
    k_msg   <<<(NEDGES + 7) / 8, 256, 0, stream>>>(edge_index, edge_type, idx_map,
                                                   comp, basis, deg, agg);
    k_finish<<<((size_t)N_ENT * DIM + 255) / 256, 256, 0, stream>>>(idx_map, root, bias, agg);

    // user path
    k_gather<<<(BUSERS * LCTX * DIM + 255) / 256, 256, 0, stream>>>(ctx, idx_map, agg,
                                                                    hbuf, BUSERS * LCTX);
    {   // tbuf = tanh(hbuf @ attn_W)
        int nt = ((BUSERS * LCTX) >> 4) * 8;
        k_gemm_wmma<<<(nt + 7) / 8, 256, 0, stream>>>(hbuf, attn_W, nullptr, tbuf,
                                                      BUSERS * LCTX, 2);
    }
    k_scores  <<<(BUSERS * LCTX + 7) / 8, 256, 0, stream>>>(tbuf, attn_b, scores);
    k_attnpool<<<BUSERS, DIM, 0, stream>>>(scores, mask, hbuf, urep);
    {
        int nt = (BUSERS >> 4) * 8;
        k_gemm_wmma<<<(nt + 7) / 8, 256, 0, stream>>>(urep, fc1_W, fc1_b, hidden, BUSERS, 1);
        k_gemm_wmma<<<(nt + 7) / 8, 256, 0, stream>>>(hidden, fc2_W, fc2_b, out_user, BUSERS, 0);
    }

    // entity path
    k_gather<<<(NEIDS * DIM + 255) / 256, 256, 0, stream>>>(eids, idx_map, agg, ereps, NEIDS);
    {
        int nt = (NEIDS >> 4) * 8;
        k_gemm_wmma<<<(nt + 7) / 8, 256, 0, stream>>>(ereps, efc1_W, efc1_b, hidden, NEIDS, 1);
        k_gemm_wmma<<<(nt + 7) / 8, 256, 0, stream>>>(hidden, efc2_W, efc2_b, out_ent, NEIDS, 0);
    }
}
